// spatialAttention_45938970198780
// MI455X (gfx1250) — matrix operand order repaired and checked by reading, not compiled, not run on hardware
//
#include <hip/hip_runtime.h>
#include <hip/hip_bf16.h>
#include <cstdint>

typedef _Float16 h16;
typedef __attribute__((ext_vector_type(16))) _Float16 v16h;
typedef __attribute__((ext_vector_type(8)))  float    v8f;

union HU { v16h v; uint32_t u[8]; };

// A-fragment (16x32, f16): lane<16 -> row=l, K{0..7} u[0..3], K{16..23} u[4..7]
//                          lane>=16 -> row=l, K{8..15},K{24..31}
static __device__ __forceinline__ v16h frag_a(const uint32_t* rowPtr, int hi) {
  HU r;
#pragma unroll
  for (int i = 0; i < 4; ++i) r.u[i]     = rowPtr[hi * 4 + i];
#pragma unroll
  for (int i = 0; i < 4; ++i) r.u[4 + i] = rowPtr[8 + hi * 4 + i];
  return r.v;
}
// B-fragment (32x16, f16): lane<16 -> col=l, K{0..7},K{16..23} ; lane>=16 -> col=l, K{8..15},K{24..31}
static __device__ __forceinline__ v16h frag_b(const uint32_t* rowPtr, int hi) {
  HU r;
#pragma unroll
  for (int i = 0; i < 4; ++i) r.u[i]     = rowPtr[hi * 4 + i];
#pragma unroll
  for (int i = 0; i < 4; ++i) r.u[4 + i] = rowPtr[8 + hi * 4 + i];
  return r.v;
}
static __device__ __forceinline__ v8f wmma_f16(v16h a, v16h b, v8f c) {
  return __builtin_amdgcn_wmma_f32_16x16x32_f16(false, a, false, b, (short)0, c,
                                                false, false);
}

// Async global -> LDS DMA (16B per lane), tracked with ASYNCcnt.
static __device__ __forceinline__ void async_b128(uint32_t lds_off,
                                                  const void* gaddr) {
  asm volatile("global_load_async_to_lds_b128 %0, %1, off"
               :
               : "v"(lds_off), "v"(gaddr)
               : "memory");
}
template <int N>
static __device__ __forceinline__ void wait_async() {
  asm volatile("s_wait_asynccnt %0" : : "n"(N) : "memory");
}
static __device__ __forceinline__ uint32_t lds_off_of(const void* p) {
  return (uint32_t)(uintptr_t)p;  // LDS aperture: low 32 bits = LDS byte offset
}

// ---------------------------------------------------------------------------
// Elementwise f32 -> f16 (weights)
// ---------------------------------------------------------------------------
__global__ __launch_bounds__(256) void f32_to_h16(const float* __restrict__ in,
                                                  h16* __restrict__ out, int n) {
  int i = blockIdx.x * 256 + threadIdx.x;
  if (i < n) out[i] = (h16)in[i];
}

// ---------------------------------------------------------------------------
// Graph bias: bias[i][j] = alpha*softmax_j(relu(E_i . E_j)) + beta*lap + mask
// ---------------------------------------------------------------------------
__global__ __launch_bounds__(256) void graph_bias(const float* __restrict__ E,
                                                  const float* __restrict__ lap,
                                                  const float* __restrict__ alpha_p,
                                                  const float* __restrict__ beta_p,
                                                  float* __restrict__ biasM) {
  const int i = blockIdx.x;  // row 0..511
  __shared__ float ei[64];
  __shared__ float tmp[512];
  __shared__ float red[8];
  __shared__ float mx_s, sum_s;
  if (threadIdx.x < 64) ei[threadIdx.x] = E[i * 64 + threadIdx.x];
  __syncthreads();
  int wave = threadIdx.x >> 5, lane = threadIdx.x & 31;
  float lmax = 0.0f;  // relu => values >= 0
  for (int j = threadIdx.x; j < 512; j += 256) {
    const float* ej = E + j * 64;
    float d = 0.0f;
#pragma unroll
    for (int c = 0; c < 64; ++c) d += ei[c] * ej[c];
    d = fmaxf(d, 0.0f);
    tmp[j] = d;
    lmax = fmaxf(lmax, d);
  }
  for (int o = 16; o > 0; o >>= 1) lmax = fmaxf(lmax, __shfl_xor(lmax, o, 32));
  if (lane == 0) red[wave] = lmax;
  __syncthreads();
  if (threadIdx.x == 0) {
    float m = red[0];
    for (int k = 1; k < 8; ++k) m = fmaxf(m, red[k]);
    mx_s = m;
  }
  __syncthreads();
  float mx = mx_s;
  float ls = 0.0f;
  for (int j = threadIdx.x; j < 512; j += 256) {
    float e = __expf(tmp[j] - mx);
    tmp[j] = e;
    ls += e;
  }
  for (int o = 16; o > 0; o >>= 1) ls += __shfl_xor(ls, o, 32);
  if (lane == 0) red[wave] = ls;
  __syncthreads();
  if (threadIdx.x == 0) {
    float s = 0.0f;
    for (int k = 0; k < 8; ++k) s += red[k];
    sum_s = s;
  }
  __syncthreads();
  float inv = 1.0f / sum_s;
  float a = alpha_p[0], bb = beta_p[0];
  for (int j = threadIdx.x; j < 512; j += 256) {
    float lp = lap[i * 512 + j];
    float m = (lp != 0.0f) ? 0.0f : -1e9f;
    biasM[i * 512 + j] = a * (tmp[j] * inv) + bb * lp + m;
  }
}

// ---------------------------------------------------------------------------
// LayerNorm row -> f16
// ---------------------------------------------------------------------------
__global__ __launch_bounds__(256) void ln_to_h16(const float* __restrict__ x,
                                                 const float* __restrict__ g,
                                                 const float* __restrict__ b,
                                                 h16* __restrict__ out, int D) {
  const int row = blockIdx.x;
  const float* xr = x + (size_t)row * D;
  __shared__ float redA[8], redB[8];
  __shared__ float mu_s, rs_s;
  int wave = threadIdx.x >> 5, lane = threadIdx.x & 31;
  float s = 0.0f, s2 = 0.0f;
  for (int c = threadIdx.x; c < D; c += 256) {
    float v = xr[c];
    s += v;
    s2 += v * v;
  }
  for (int o = 16; o > 0; o >>= 1) {
    s += __shfl_xor(s, o, 32);
    s2 += __shfl_xor(s2, o, 32);
  }
  if (lane == 0) { redA[wave] = s; redB[wave] = s2; }
  __syncthreads();
  if (threadIdx.x == 0) {
    float ts = 0.0f, ts2 = 0.0f;
    for (int k = 0; k < 8; ++k) { ts += redA[k]; ts2 += redB[k]; }
    float mu = ts / D;
    mu_s = mu;
    rs_s = rsqrtf(ts2 / D - mu * mu + 1e-5f);
  }
  __syncthreads();
  float mu = mu_s, rs = rs_s;
  for (int c = threadIdx.x; c < D; c += 256)
    out[(size_t)row * D + c] = (h16)((xr[c] - mu) * rs * g[c] + b[c]);
}

// ---------------------------------------------------------------------------
// Tiled GEMM: C[M,N] = A[M,Kd] @ W[N,Kd]^T + bvec, WMMA f16->f32.
// Double-buffered LDS tiles filled by GLOBAL_LOAD_ASYNC_TO_LDS_B128 (ASYNCcnt),
// so tile k+1's HBM traffic overlaps tile k's WMMAs with zero staging VGPRs.
// MODE 0: out f16           MODE 1: out f16 scattered to V^T per head
// MODE 2: out f32 = resid+C MODE 3: out f16 = relu(C)
// Block 128x128, BK=32, 8 wave32s (2x4), wave tile 64x32 (4x2 frags).
// ---------------------------------------------------------------------------
template <int MODE>
__global__ __launch_bounds__(256) void gemm_awt(const h16* __restrict__ A,
                                                const h16* __restrict__ W,
                                                const float* __restrict__ bvec,
                                                const float* __restrict__ resid,
                                                h16* __restrict__ outH,
                                                float* __restrict__ outF,
                                                int M, int N, int Kd) {
  __shared__ uint32_t As[2][128 * 20];  // 128 rows x 40 halves (pad 8), x2 buf
  __shared__ uint32_t Bs[2][128 * 20];
  const int t = threadIdx.x;
  const int lane = t & 31, wave = t >> 5;
  const int wr = wave >> 2, wc = wave & 3;
  const int m0 = blockIdx.y * 128, n0 = blockIdx.x * 128;
  const int l = lane & 15, hi = lane >> 4;

  v8f acc[4][2];
#pragma unroll
  for (int i = 0; i < 4; ++i)
#pragma unroll
    for (int j = 0; j < 2; ++j)
#pragma unroll
      for (int e = 0; e < 8; ++e) acc[i][j][e] = 0.0f;

  // Per-thread DMA slice: 2 threads per 32-half row; 32 B (2 x b128) each,
  // for A and for B => 4 async b128 per thread per K-tile.
  const int lr = t >> 1, lc = t & 1;
  const h16* gA = A + (size_t)(m0 + lr) * Kd + lc * 16;
  const h16* gB = W + (size_t)(n0 + lr) * Kd + lc * 16;
  const uint32_t ldsA = lds_off_of(&As[0][lr * 20 + lc * 8]);
  const uint32_t ldsB = lds_off_of(&Bs[0][lr * 20 + lc * 8]);
  const uint32_t bufB = 128 * 20 * 4;  // bytes between buffer 0 and 1

  auto issue = [&](int buf, int k0) {
    async_b128(ldsA + buf * bufB, gA + k0);
    async_b128(ldsA + buf * bufB + 16, gA + k0 + 8);
    async_b128(ldsB + buf * bufB, gB + k0);
    async_b128(ldsB + buf * bufB + 16, gB + k0 + 8);
  };

  const int T = Kd >> 5;
  issue(0, 0);
  for (int k = 0; k < T; ++k) {
    const int cur = k & 1;
    if (k + 1 < T) {
      issue(cur ^ 1, (k + 1) << 5);
      wait_async<4>();  // the 4 b128s for buffer `cur` have landed
    } else {
      wait_async<0>();
    }
    __syncthreads();  // all waves' DMA for `cur` visible

    const uint32_t* Ab = As[cur];
    const uint32_t* Bb = Bs[cur];
    v16h bf[2];
#pragma unroll
    for (int j = 0; j < 2; ++j)
      bf[j] = frag_b(Bb + (wc * 32 + j * 16 + l) * 20, hi);
#pragma unroll
    for (int i = 0; i < 4; ++i) {
      v16h af = frag_a(Ab + (wr * 64 + i * 16 + l) * 20, hi);
#pragma unroll
      for (int j = 0; j < 2; ++j) acc[i][j] = wmma_f16(af, bf[j], acc[i][j]);
    }
    __syncthreads();  // safe to overwrite `cur` in iteration k+1
  }

#pragma unroll
  for (int i = 0; i < 4; ++i)
#pragma unroll
    for (int j = 0; j < 2; ++j)
#pragma unroll
      for (int e = 0; e < 8; ++e) {
        int row = m0 + wr * 64 + i * 16 + (hi ? 8 + e : e);
        int col = n0 + wc * 32 + j * 16 + l;
        float v = acc[i][j][e] + bvec[col];
        if (MODE == 0) {
          outH[(size_t)row * N + col] = (h16)v;
        } else if (MODE == 1) {  // scatter into per-head V^T [BT*K*dh, 512]
          int bt = row >> 9, n = row & 511;
          int kh = col >> 6, e2 = col & 63;
          outH[(size_t)((bt * 8 + kh) * 64 + e2) * 512 + n] = (h16)v;
        } else if (MODE == 2) {
          size_t idx = (size_t)row * N + col;
          outF[idx] = resid[idx] + v;
        } else {  // 3
          outH[(size_t)row * N + col] = (h16)fmaxf(v, 0.0f);
        }
      }
}

// ---------------------------------------------------------------------------
// Fused attention: per (head, 32-query-row tile):
//   S = (Q Kh^T)/8 + biasM ; P = softmax_row(S) ; O = P V  (V pre-transposed)
// ---------------------------------------------------------------------------
__global__ __launch_bounds__(256) void attn_kernel(const h16* __restrict__ Q,
                                                   const h16* __restrict__ Kh,
                                                   const h16* __restrict__ Vt,
                                                   const float* __restrict__ biasM,
                                                   h16* __restrict__ O) {
  extern __shared__ char smem[];
  float* S = (float*)smem;                                   // [32][512] 64 KB
  h16* P = (h16*)(smem + 32 * 512 * 4);                      // [32][512] 32 KB
  float* rowsum = (float*)(smem + 32 * 512 * 4 + 32 * 512 * 2);  // [32]

  const int head = blockIdx.y;  // 0..255 = bt*8 + kh
  const int tile = blockIdx.x;  // 0..15
  const int bt = head >> 3, kh = head & 7;
  const int m0 = tile * 32;  // query-node base
  const int t = threadIdx.x, lane = t & 31, wave = t >> 5;
  const int l = lane & 15, hi = lane >> 4;
  const size_t hb = ((size_t)bt * 512) * 512 + (size_t)kh * 64;

  // ---- Stage 1: S (each wave: 32 rows x 64 key-cols) ----
  {
    v8f acc[2][4];
#pragma unroll
    for (int i = 0; i < 2; ++i)
#pragma unroll
      for (int j = 0; j < 4; ++j)
#pragma unroll
        for (int e = 0; e < 8; ++e) acc[i][j][e] = 0.0f;
#pragma unroll
    for (int kk = 0; kk < 2; ++kk) {  // dh = 64 -> 2 k-steps
      v16h af[2];
#pragma unroll
      for (int i = 0; i < 2; ++i)
        af[i] = frag_a(
            (const uint32_t*)(Q + hb + (size_t)(m0 + i * 16 + l) * 512 + kk * 32),
            hi);
#pragma unroll
      for (int j = 0; j < 4; ++j) {
        int kn = wave * 64 + j * 16 + l;  // key node
        v16h bf = frag_b(
            (const uint32_t*)(Kh + hb + (size_t)kn * 512 + kk * 32), hi);
#pragma unroll
        for (int i = 0; i < 2; ++i) acc[i][j] = wmma_f16(af[i], bf, acc[i][j]);
      }
    }
#pragma unroll
    for (int i = 0; i < 2; ++i)
#pragma unroll
      for (int j = 0; j < 4; ++j)
#pragma unroll
        for (int e = 0; e < 8; ++e) {
          int r = i * 16 + (hi ? 8 + e : e);
          int c = wave * 64 + j * 16 + l;
          S[r * 512 + c] =
              acc[i][j][e] * 0.125f + biasM[(size_t)(m0 + r) * 512 + c];
        }
  }
  __syncthreads();

  // ---- Stage 2: row softmax (wave handles 4 rows) ----
  for (int rr = 0; rr < 4; ++rr) {
    int r = wave * 4 + rr;
    float vals[16];
    float mx = -3.4e38f;
#pragma unroll
    for (int ii = 0; ii < 16; ++ii) {
      float v = S[r * 512 + lane + 32 * ii];
      vals[ii] = v;
      mx = fmaxf(mx, v);
    }
    for (int o = 16; o > 0; o >>= 1) mx = fmaxf(mx, __shfl_xor(mx, o, 32));
    float ss = 0.0f;
#pragma unroll
    for (int ii = 0; ii < 16; ++ii) {
      float e = __expf(vals[ii] - mx);
      ss += e;
      P[r * 512 + lane + 32 * ii] = (h16)e;
    }
    for (int o = 16; o > 0; o >>= 1) ss += __shfl_xor(ss, o, 32);
    if (lane == 0) rowsum[r] = ss;
  }
  __syncthreads();

  // ---- Stage 3: O = P @ V (wave: mf=rows16, nf=cols16; K=512 -> 16 steps) ----
  {
    const int mf = wave >> 2, nf = wave & 3;
    v8f acc;
#pragma unroll
    for (int e = 0; e < 8; ++e) acc[e] = 0.0f;
    const size_t vbase = (size_t)((bt * 8 + kh) * 64) * 512;
#pragma unroll
    for (int kk = 0; kk < 16; ++kk) {
      v16h af = frag_a((const uint32_t*)(P + (mf * 16 + l) * 512 + kk * 32), hi);
      v16h bf = frag_b(
          (const uint32_t*)(Vt + vbase + (size_t)(nf * 16 + l) * 512 + kk * 32),
          hi);
      acc = wmma_f16(af, bf, acc);
    }
#pragma unroll
    for (int e = 0; e < 8; ++e) {
      int r = mf * 16 + (hi ? 8 + e : e);
      int c = nf * 16 + l;
      float v = acc[e] / rowsum[r];
      O[(size_t)(bt * 512 + m0 + r) * 512 + (kh * 64 + c)] = (h16)v;
    }
  }
}

// ---------------------------------------------------------------------------
extern "C" void kernel_launch(void* const* d_in, const int* in_sizes, int n_in,
                              void* d_out, int out_size, void* d_ws,
                              size_t ws_size, hipStream_t stream) {
  const float* x = (const float*)d_in[0];
  const float* lap = (const float*)d_in[1];
  const float* emb = (const float*)d_in[2];
  const float* Wq = (const float*)d_in[3];
  const float* bq = (const float*)d_in[4];
  const float* Wk = (const float*)d_in[5];
  const float* bk = (const float*)d_in[6];
  const float* Wv = (const float*)d_in[7];
  const float* bv = (const float*)d_in[8];
  const float* g1 = (const float*)d_in[9];
  const float* be1 = (const float*)d_in[10];
  const float* g2 = (const float*)d_in[11];
  const float* be2 = (const float*)d_in[12];
  const float* alpha = (const float*)d_in[13];
  const float* beta = (const float*)d_in[14];
  const float* Wo = (const float*)d_in[15];
  const float* bo = (const float*)d_in[16];
  const float* W1 = (const float*)d_in[17];
  const float* b1 = (const float*)d_in[18];
  const float* W2 = (const float*)d_in[19];
  const float* b2 = (const float*)d_in[20];

  char* ws = (char*)d_ws;
  size_t off = 0;
  auto alloc = [&](size_t bytes) -> void* {
    void* p = ws + off;
    off += (bytes + 255) & ~(size_t)255;
    return p;
  };
  const size_t R = 16384;  // B*T*N
  h16* xn = (h16*)alloc(R * 512 * 2);
  h16* qh = (h16*)alloc(R * 512 * 2);
  h16* khb = (h16*)alloc(R * 512 * 2);
  h16* vt = (h16*)alloc(R * 512 * 2);
  h16* ao = (h16*)alloc(R * 512 * 2);
  float* x1 = (float*)alloc(R * 512 * 4);
  h16* x1n = (h16*)alloc(R * 512 * 2);
  h16* hh = (h16*)alloc(R * 2048 * 2);
  float* bm = (float*)alloc(512 * 512 * 4);
  h16* wqh = (h16*)alloc(512 * 512 * 2);
  h16* wkh = (h16*)alloc(512 * 512 * 2);
  h16* wvh = (h16*)alloc(512 * 512 * 2);
  h16* woh = (h16*)alloc(512 * 512 * 2);
  h16* w1h = (h16*)alloc(2048 * 512 * 2);
  h16* w2h = (h16*)alloc(512 * 2048 * 2);

  const int nsq = 512 * 512, nbig = 2048 * 512;
  f32_to_h16<<<(nsq + 255) / 256, 256, 0, stream>>>(Wq, wqh, nsq);
  f32_to_h16<<<(nsq + 255) / 256, 256, 0, stream>>>(Wk, wkh, nsq);
  f32_to_h16<<<(nsq + 255) / 256, 256, 0, stream>>>(Wv, wvh, nsq);
  f32_to_h16<<<(nsq + 255) / 256, 256, 0, stream>>>(Wo, woh, nsq);
  f32_to_h16<<<(nbig + 255) / 256, 256, 0, stream>>>(W1, w1h, nbig);
  f32_to_h16<<<(nbig + 255) / 256, 256, 0, stream>>>(W2, w2h, nbig);

  graph_bias<<<512, 256, 0, stream>>>(emb, lap, alpha, beta, bm);
  ln_to_h16<<<16384, 256, 0, stream>>>(x, g1, be1, xn, 512);

  dim3 g512(512 / 128, 16384 / 128);
  gemm_awt<0><<<g512, 256, 0, stream>>>(xn, wqh, bq, nullptr, qh, nullptr,
                                        16384, 512, 512);
  gemm_awt<0><<<g512, 256, 0, stream>>>(xn, wkh, bk, nullptr, khb, nullptr,
                                        16384, 512, 512);
  gemm_awt<1><<<g512, 256, 0, stream>>>(xn, wvh, bv, nullptr, vt, nullptr,
                                        16384, 512, 512);

  const int attn_lds = 32 * 512 * 4 + 32 * 512 * 2 + 32 * 4;  // 98432 B
  hipFuncSetAttribute((const void*)attn_kernel,
                      hipFuncAttributeMaxDynamicSharedMemorySize, attn_lds);
  attn_kernel<<<dim3(16, 256), 256, attn_lds, stream>>>(qh, khb, vt, bm, ao);

  gemm_awt<2><<<g512, 256, 0, stream>>>(ao, woh, bo, x, nullptr, x1, 16384, 512,
                                        512);
  ln_to_h16<<<16384, 256, 0, stream>>>(x1, g2, be2, x1n, 512);
  gemm_awt<3><<<dim3(2048 / 128, 128), 256, 0, stream>>>(
      x1n, w1h, b1, nullptr, hh, nullptr, 16384, 2048, 512);
  gemm_awt<2><<<g512, 256, 0, stream>>>(hh, w2h, b2, x1, nullptr, (float*)d_out,
                                        16384, 512, 2048);
}
